// RGCNLayer_26190710571675
// MI455X (gfx1250) — matrix operand order repaired and checked by reading, not compiled, not run on hardware
//
#include <hip/hip_runtime.h>
#include <hip/hip_bf16.h>

// ---------------------------------------------------------------------------
// RGCN layer for MI455X (gfx1250), wave32 + WMMA f16 path.
//
// Pipeline:
//   0) memset agg + channel sums to zero
//   1) cvt_x     : x (f32) -> xh (f16)
//   2) pack_b    : [w_loop | w_bases] (f32) -> WMMA B-fragment-packed f16
//   3) gemm      : LDS-staged A (128 rows, bank-conflict-padded), B fragments
//                  preloaded + amortized over 8 M-tiles,
//                  v_wmma_f32_16x16x32_f16 with f32 accumulation.
//                  Writes loopmsg (f32) + hb (f16).
//   4) edge      : gather hb[src] (f16, 102 MB -> L2-resident), combine with
//                  w_rel coefs, scatter f32 atomics into agg[dst]
//   5) stats     : node = agg*norm + bias + loop; LDS-reduce sum/sumsq,
//                  atomics into 256 channel accumulators
//   6) final     : batchnorm + relu -> d_out
// ---------------------------------------------------------------------------

typedef __attribute__((ext_vector_type(16))) _Float16 v16h;
typedef __attribute__((ext_vector_type(8)))  _Float16 v8h;
typedef __attribute__((ext_vector_type(4)))  _Float16 v4h;
typedef __attribute__((ext_vector_type(8)))  float    v8f;

#define N_NODES 50000
#define N_EDGES 600000
#define IN_DIM  128
#define OUT_DIM 128
#define N_REL   100
#define N_BASES 8
#define O_TOTAL (OUT_DIM + N_BASES * OUT_DIM)   // 1152 = 72 n-tiles
#define N_TILES_N (O_TOTAL / 16)                // 72
#define N_TILES_M (N_NODES / 16)                // 3125
#define MT_PER_BLOCK 8                          // M-tiles per block (128 rows)
#define GEMM_BLOCKS ((N_TILES_M + MT_PER_BLOCK - 1) / MT_PER_BLOCK)  // 391
#define K_STEPS  (IN_DIM / 32)                  // 4
#define XLDS_STRIDE 136                         // 128 + 8 halfs: 4-bank row skew
#define BN_EPS 1e-5f

static __device__ __forceinline__ void atomicAddF(float* p, float v) {
    unsafeAtomicAdd(p, v);   // global_atomic_add_f32 on gfx1250
}

// ---- 1) convert x to f16 ---------------------------------------------------
__global__ void cvt_x_kernel(const float* __restrict__ x, _Float16* __restrict__ xh, int n) {
    int i = blockIdx.x * 256 + threadIdx.x;
    if (i < n) xh[i] = (_Float16)x[i];
}

// ---- 2) pack B = [w_loop | w_bases] into WMMA B-fragment layout ------------
// bpack[((ks*72 + nt)*32 + lane)*16 + j] = B[k][n] with
//   k = ks*32 + (j<8 ? (lane>>4)*8 + j : 16 + (lane>>4)*8 + (j-8)),  n = nt*16 + (lane&15)
__global__ void pack_b_kernel(const float* __restrict__ wloop,
                              const float* __restrict__ wbases,
                              _Float16* __restrict__ bpack) {
    int idx = blockIdx.x * 256 + threadIdx.x;
    if (idx >= K_STEPS * N_TILES_N * 32 * 16) return;
    int j    = idx & 15;
    int lane = (idx >> 4) & 31;
    int rest = idx >> 9;
    int nt   = rest % N_TILES_N;
    int ks   = rest / N_TILES_N;
    int k = ks * 32 + (j < 8 ? (lane >> 4) * 8 + j : 16 + (lane >> 4) * 8 + (j - 8));
    int n = nt * 16 + (lane & 15);
    float v;
    if (n < OUT_DIM) {
        v = wloop[k * OUT_DIM + n];
    } else {
        int oc = n - OUT_DIM;
        int b = oc >> 7, o = oc & 127;
        v = wbases[((size_t)b * IN_DIM + k) * OUT_DIM + o];
    }
    bpack[idx] = (_Float16)v;
}

// ---- 3) fused GEMM: [50000x128] @ [128x1152] via WMMA ----------------------
// grid = 391 blocks x 256 threads (8 waves). Block stages 128 A-rows in LDS;
// wave w handles n-tiles w, w+8, ... (9 each), 8 M-tiles per n-tile.
__global__ __launch_bounds__(256) void
gemm_kernel(const _Float16* __restrict__ xh,
            const _Float16* __restrict__ bpack,
            _Float16* __restrict__ hb,
            float* __restrict__ loopmsg) {
    __shared__ _Float16 xlds[128 * XLDS_STRIDE];   // ~34 KB, row-skewed

    const int mtBase = blockIdx.x * MT_PER_BLOCK;
    const int nmt    = min(MT_PER_BLOCK, N_TILES_M - mtBase);  // uniform
    const int t    = threadIdx.x;
    const int wave = t >> 5;
    const int lane = t & 31;
    const int rowInTile = lane & 15;
    const int khalf = (lane >> 4) ? 8 : 0;   // A layout: hi-lanes hold K+8 chunk
    const int rbase = (lane >> 4) ? 8 : 0;   // C/D layout: hi-lanes = M+8

    // Stage A rows (row-major, stride 136 halfs) — thread t copies half a row.
    {
        int r = t >> 1;                      // 0..127
        int half = t & 1;                    // 64-half chunk
        if (r < nmt * 16) {
            const _Float16* srcp = xh + (size_t)(mtBase * 16 + r) * IN_DIM + half * 64;
            _Float16* dstl = xlds + r * XLDS_STRIDE + half * 64;
#pragma unroll
            for (int c = 0; c < 8; ++c)
                *(v8h*)(dstl + c * 8) = *(const v8h*)(srcp + c * 8);
        }
    }
    __syncthreads();

    for (int nt = wave; nt < N_TILES_N; nt += 8) {
        // Preload all 4 B fragments (8 x b128 global loads clause together).
        v16h bfrag[K_STEPS];
#pragma unroll
        for (int ks = 0; ks < K_STEPS; ++ks)
            bfrag[ks] = *(const v16h*)(bpack +
                ((size_t)((ks * N_TILES_N + nt) * 32 + lane) << 4));

        const int ncol = nt * 16 + rowInTile;
#pragma unroll
        for (int mi = 0; mi < MT_PER_BLOCK; ++mi) {
            if (mi < nmt) {                       // uniform guard, EXEC full
                const _Float16* arow = xlds + (mi * 16 + rowInTile) * XLDS_STRIDE;
                v8f acc = {};
#pragma unroll
                for (int ks = 0; ks < K_STEPS; ++ks) {
                    int kb = ks * 32 + khalf;
                    v8h lo = *(const v8h*)(arow + kb);        // K = kb .. kb+7
                    v8h hi = *(const v8h*)(arow + kb + 16);   // K = kb+16 .. kb+23
                    v16h a;
#pragma unroll
                    for (int j = 0; j < 8; ++j) { a[j] = lo[j]; a[8 + j] = hi[j]; }
                    acc = __builtin_amdgcn_wmma_f32_16x16x32_f16(
                              false, a, false, bfrag[ks], (short)0, acc,
                              false, false);
                }
                const int m0 = (mtBase + mi) * 16;
                if (ncol < OUT_DIM) {             // w_loop columns -> f32
#pragma unroll
                    for (int r = 0; r < 8; ++r)
                        loopmsg[(size_t)(m0 + rbase + r) * OUT_DIM + ncol] = acc[r];
                } else {                          // basis columns -> f16 hb
                    int oc = ncol - OUT_DIM;      // hb[row][b][o] = row*1024+oc
#pragma unroll
                    for (int r = 0; r < 8; ++r)
                        hb[(size_t)(m0 + rbase + r) * (N_BASES * OUT_DIM) + oc] =
                            (_Float16)acc[r];
                }
            }
        }
    }
}

// ---- 4) per-edge gather/combine/scatter ------------------------------------
// One wave per edge; lane handles channels [4*lane, 4*lane+4).
__global__ void edge_kernel(const _Float16* __restrict__ hb,
                            const float* __restrict__ wrel,
                            const int* __restrict__ src,
                            const int* __restrict__ dst,
                            const int* __restrict__ eid,
                            float* __restrict__ agg) {
    const int wave = threadIdx.x >> 5;
    const int lane = threadIdx.x & 31;
    const int e = blockIdx.x * 8 + wave;
    if (e >= N_EDGES) return;
    const int s = src[e], d = dst[e], r = eid[e];
    const float* cf = wrel + (size_t)r * N_BASES;
    const _Float16* h = hb + (size_t)s * (N_BASES * OUT_DIM) + lane * 4;
    float a0 = 0.f, a1 = 0.f, a2 = 0.f, a3 = 0.f;
#pragma unroll
    for (int b = 0; b < N_BASES; ++b) {
        float c = cf[b];
        v4h hv = *(const v4h*)(h + b * OUT_DIM);
        a0 += c * (float)hv[0];
        a1 += c * (float)hv[1];
        a2 += c * (float)hv[2];
        a3 += c * (float)hv[3];
    }
    float* ap = agg + (size_t)d * OUT_DIM + lane * 4;
    atomicAddF(ap + 0, a0);
    atomicAddF(ap + 1, a1);
    atomicAddF(ap + 2, a2);
    atomicAddF(ap + 3, a3);
}

// ---- 5) batchnorm statistics ----------------------------------------------
// Block handles 32 node rows; thread t -> channel t&127, half t>>7.
__global__ void stats_kernel(const float* __restrict__ agg,
                             const float* __restrict__ norm,
                             const float* __restrict__ wbias,
                             const float* __restrict__ loopmsg,
                             float* __restrict__ sums) {   // sums[0:128]=S, [128:256]=S2
    __shared__ float l1[256];
    __shared__ float l2[256];
    const int t = threadIdx.x;
    const int c = t & 127, h = t >> 7;
    const int base = blockIdx.x * 32;
    float s1 = 0.f, s2 = 0.f;
#pragma unroll 4
    for (int i = 0; i < 16; ++i) {
        int row = base + h + 2 * i;
        if (row < N_NODES) {
            float v = agg[(size_t)row * OUT_DIM + c] * norm[row] + wbias[c] +
                      loopmsg[(size_t)row * OUT_DIM + c];
            s1 += v;
            s2 += v * v;
        }
    }
    l1[t] = s1; l2[t] = s2;
    __syncthreads();
    if (t < 128) {
        atomicAddF(&sums[c],       l1[t] + l1[t + 128]);
        atomicAddF(&sums[128 + c], l2[t] + l2[t + 128]);
    }
}

// ---- 6) batchnorm apply + relu --------------------------------------------
__global__ void final_kernel(const float* __restrict__ agg,
                             const float* __restrict__ norm,
                             const float* __restrict__ wbias,
                             const float* __restrict__ loopmsg,
                             const float* __restrict__ sums,
                             const float* __restrict__ gamma,
                             const float* __restrict__ beta,
                             float* __restrict__ out) {
    const int t = threadIdx.x;
    const int c = t & 127;
    const int row = blockIdx.x * 2 + (t >> 7);
    if (row >= N_NODES) return;
    const float invN = 1.0f / (float)N_NODES;
    float mean = sums[c] * invN;
    float var  = sums[128 + c] * invN - mean * mean;
    float v = agg[(size_t)row * OUT_DIM + c] * norm[row] + wbias[c] +
              loopmsg[(size_t)row * OUT_DIM + c];
    float y = gamma[c] * (v - mean) * rsqrtf(var + BN_EPS) + beta[c];
    out[(size_t)row * OUT_DIM + c] = fmaxf(y, 0.f);
}

// ---------------------------------------------------------------------------
extern "C" void kernel_launch(void* const* d_in, const int* in_sizes, int n_in,
                              void* d_out, int out_size, void* d_ws, size_t ws_size,
                              hipStream_t stream) {
    const float* x      = (const float*)d_in[0];
    const float* norm   = (const float*)d_in[1];
    const float* wloop  = (const float*)d_in[2];
    const float* wbases = (const float*)d_in[3];
    const float* wrel   = (const float*)d_in[4];
    const float* wbias  = (const float*)d_in[5];
    const float* gamma  = (const float*)d_in[6];
    const float* beta   = (const float*)d_in[7];
    const int*   src    = (const int*)d_in[8];
    const int*   dst    = (const int*)d_in[9];
    const int*   eid    = (const int*)d_in[10];
    float* out = (float*)d_out;

    // Workspace layout (256B aligned slices)
    char* ws = (char*)d_ws;
    size_t off = 0;
    auto take = [&](size_t bytes) {
        void* p = ws + off;
        off += (bytes + 255) & ~(size_t)255;
        return p;
    };
    _Float16* xh      = (_Float16*)take((size_t)N_NODES * IN_DIM * sizeof(_Float16));          // 12.8 MB
    _Float16* bpack   = (_Float16*)take((size_t)K_STEPS * N_TILES_N * 32 * 16 * sizeof(_Float16)); // 0.3 MB
    _Float16* hb      = (_Float16*)take((size_t)N_NODES * N_BASES * OUT_DIM * sizeof(_Float16));   // 102.4 MB (L2-resident)
    float*    loopmsg = (float*)take((size_t)N_NODES * OUT_DIM * sizeof(float));               // 25.6 MB
    float*    agg     = (float*)take((size_t)N_NODES * OUT_DIM * sizeof(float));               // 25.6 MB
    float*    sums    = (float*)take(256 * sizeof(float));

    // 0) zero accumulators (graph-capture-safe async memsets)
    hipMemsetAsync(agg,  0, (size_t)N_NODES * OUT_DIM * sizeof(float), stream);
    hipMemsetAsync(sums, 0, 256 * sizeof(float), stream);

    // 1) x -> f16
    {
        int n = N_NODES * IN_DIM;
        cvt_x_kernel<<<(n + 255) / 256, 256, 0, stream>>>(x, xh, n);
    }
    // 2) pack weights into WMMA B-fragment layout
    {
        int n = K_STEPS * N_TILES_N * 32 * 16;
        pack_b_kernel<<<(n + 255) / 256, 256, 0, stream>>>(wloop, wbases, bpack);
    }
    // 3) fused WMMA GEMM: loop message + 8 basis projections
    gemm_kernel<<<GEMM_BLOCKS, 256, 0, stream>>>(xh, bpack, hb, loopmsg);

    // 4) edge gather/combine/scatter (one wave per edge)
    edge_kernel<<<(N_EDGES + 7) / 8, 256, 0, stream>>>(hb, wrel, src, dst, eid, agg);

    // 5) batchnorm stats
    stats_kernel<<<(N_NODES + 31) / 32, 256, 0, stream>>>(agg, norm, wbias, loopmsg, sums);

    // 6) batchnorm apply + relu
    final_kernel<<<(N_NODES + 1) / 2, 256, 0, stream>>>(agg, norm, wbias, loopmsg, sums,
                                                        gamma, beta, out);
    (void)in_sizes; (void)n_in; (void)out_size; (void)ws_size;
}
